// PanguModel_59339268161775
// MI455X (gfx1250) — matrix operand order repaired and checked by reading, not compiled, not run on hardware
//
#include <hip/hip_runtime.h>

#define NW     960
#define NT     144
#define DIMM   192
#define HEADS  6
#define HD     32
#define QKVN   576   // 3*DIM
#define MROWS  (NW * NT)          // 138240
#define PSTR   160                // padded token stride (144 -> 160) for PV k-loop
#define MB_W   5                  // mask bit-words per row (160 bits)

typedef __attribute__((ext_vector_type(16))) __bf16 v16bf;
typedef __attribute__((ext_vector_type(8)))  float  v8f;

union FragU { uint4 u[2]; v16bf v; };

__device__ __forceinline__ unsigned short f2bf(float f) {
  unsigned int x = __float_as_uint(f);
  x += 0x7fffu + ((x >> 16) & 1u);            // round-to-nearest-even
  return (unsigned short)(x >> 16);
}

// A fragment (16x32 bf16) per ISA layout: lane(0..15)=rows, halves hold
// K = kb..kb+7 and kb+16..kb+23 with kb = (lane<16)?0:8.
__device__ __forceinline__ v16bf load_a_frag(const unsigned short* base, int stride, int lane) {
  const unsigned short* p = base + (lane & 15) * stride + ((lane < 16) ? 0 : 8);
  FragU f;
  f.u[0] = *reinterpret_cast<const uint4*>(p);
  f.u[1] = *reinterpret_cast<const uint4*>(p + 16);
  return f.v;
}

// B fragment (32x16 bf16) from Bt (N x K row-major): lane column N = lane&15,
// K = kb..kb+7 and kb+16..kb+23 with kb = (lane<16)?0:8.
__device__ __forceinline__ v16bf load_b_frag(const unsigned short* base, int stride, int lane) {
  const unsigned short* p = base + (lane & 15) * stride + ((lane < 16) ? 0 : 8);
  FragU f;
  f.u[0] = *reinterpret_cast<const uint4*>(p);
  f.u[1] = *reinterpret_cast<const uint4*>(p + 16);
  return f.v;
}

// ---------------- precompute kernels ----------------

__global__ __launch_bounds__(256) void f32_to_bf16_kernel(const float* __restrict__ in,
                                                          unsigned short* __restrict__ out,
                                                          long long n) {
  long long i = (long long)blockIdx.x * 256 + threadIdx.x;
  if (i < n) out[i] = f2bf(in[i]);
}

// Wt[n*K + k] = bf16(W[k*N + n])
__global__ __launch_bounds__(256) void transpose_to_bf16(const float* __restrict__ W,
                                                         unsigned short* __restrict__ Wt,
                                                         int K, int N) {
  int i = blockIdx.x * 256 + threadIdx.x;
  if (i < K * N) {
    int k = i / N, n = i % N;
    Wt[n * K + k] = f2bf(W[i]);
  }
}

// bias_full[h][n][m] = bias_table[rel_index[n][m]][h]
__global__ __launch_bounds__(256) void bias_expand(const float* __restrict__ table,
                                                   const int* __restrict__ rel,
                                                   float* __restrict__ out) {
  int i = blockIdx.x * 256 + threadIdx.x;
  const int tot = HEADS * NT * NT;
  if (i < tot) {
    int h  = i / (NT * NT);
    int rm = i % (NT * NT);
    out[i] = table[rel[rm] * HEADS + h];
  }
}

// compress int32 mask to 1 bit/element via wave32 ballot:
// mb[w*NT + n][j] bit b = (mask[w][n][j*32+b] != 0); 5 words/row (160 bits, padded)
__global__ __launch_bounds__(256) void mask_to_bits(const int* __restrict__ mask,
                                                    unsigned int* __restrict__ mb) {
  long long i = (long long)blockIdx.x * 256 + threadIdx.x;   // over NW*NT*160 (exact)
  long long row = i / (MB_W * 32);
  int m = (int)(i % (MB_W * 32));
  int pred = (m < NT) ? (mask[row * NT + m] != 0) : 0;
  unsigned long long bal = __ballot(pred);
  if ((threadIdx.x & 31) == 0) mb[row * MB_W + (m >> 5)] = (unsigned int)bal;
}

// ---------------- WMMA GEMM: C = A(MxK) @ Bt^T + bias ----------------
// Register-blocked: one wave computes a 16x64 strip (4 accumulators),
// so each A fragment feeds 4 WMMAs per k-step.

template <bool OUT_F32>
__global__ __launch_bounds__(128) void gemm_bf16(const unsigned short* __restrict__ A,
                                                 const unsigned short* __restrict__ Bt,
                                                 const float* __restrict__ bias,
                                                 void* __restrict__ Cout,
                                                 int M, int N, int K) {
  int lane = threadIdx.x & 31;
  int wid  = threadIdx.x >> 5;
  long long tile = (long long)blockIdx.x * 4 + wid;
  int nstrips = N >> 6;                       // 64-col strips
  long long mt = tile / nstrips;
  int ns = (int)(tile % nstrips);
  if (mt >= (M >> 4)) return;

  const unsigned short* arow = A  + (mt * 16) * (long long)K;
  const unsigned short* brow = Bt + ((long long)ns * 64) * (long long)K;

  v8f acc[4] = {};
  for (int kk = 0; kk < K; kk += 32) {
    v16bf a = load_a_frag(arow + kk, K, lane);
    if (kk + 32 < K)                          // stream-prefetch next A k-slice
      __builtin_prefetch(arow + (lane & 15) * K + kk + 32, 0, 0);
#pragma unroll
    for (int j = 0; j < 4; ++j) {
      v16bf b = load_b_frag(brow + (long long)j * 16 * K + kk, K, lane);
      acc[j] = __builtin_amdgcn_wmma_f32_16x16x32_bf16(false, a, false, b, (short)0, acc[j], false, false);
    }
  }

  long long row0 = mt * 16 + ((lane < 16) ? 0 : 8);
#pragma unroll
  for (int j = 0; j < 4; ++j) {
    int col = ns * 64 + j * 16 + (lane & 15);
    float bv = bias[col];
    if (OUT_F32) {
      float* C = (float*)Cout;
#pragma unroll
      for (int r = 0; r < 8; ++r)
        C[(row0 + r) * (long long)N + col] = acc[j][r] + bv;
    } else {
      unsigned short* C = (unsigned short*)Cout;
#pragma unroll
      for (int r = 0; r < 8; ++r)
        C[(row0 + r) * (long long)N + col] = f2bf(acc[j][r] + bv);
    }
  }
}

// ---------------- fused windowed attention, one block per (window, head) ----------------

__global__ __launch_bounds__(256) void attn_kernel(const unsigned short* __restrict__ qkv, // MROWS x 576 bf16
                                                   const unsigned int* __restrict__ maskb, // NW*NT x 5 bit-words
                                                   const float* __restrict__ bias_full,    // HEADS x NT x NT
                                                   unsigned short* __restrict__ attn_out)  // MROWS x DIM bf16
{
  __shared__ unsigned short Ksh[NT * HD];        //  9 KB: K rows (bf16)
  __shared__ unsigned short Vt[HD * PSTR];       // 10 KB: V transposed, token-padded zeros
  __shared__ unsigned short Psh[8 * 16 * PSTR];  // 40 KB: per-wave P row-blocks
  __shared__ unsigned int   Mbsh[NT * MB_W];     // 2.8 KB: window mask bits

  int w = blockIdx.x / HEADS;
  int h = blockIdx.x % HEADS;
  int tid = threadIdx.x;
  int lane = tid & 31;
  int wid  = tid >> 5;
  const long long rowbase = (long long)w * NT;

  // cooperative load K (144x32 bf16, b128 chunks)
  for (int i = tid; i < NT * 4; i += 256) {
    int r = i >> 2, q = i & 3;
    const uint4* src = reinterpret_cast<const uint4*>(qkv + (rowbase + r) * QKVN + DIMM + h * HD + q * 8);
    *reinterpret_cast<uint4*>(&Ksh[r * HD + q * 8]) = *src;
  }
  // V transposed into LDS: Vt[c][n] = V[n][c]
  for (int i = tid; i < NT * HD; i += 256) {
    int n = i >> 5, c = i & 31;
    Vt[c * PSTR + n] = qkv[(rowbase + n) * QKVN + 2 * DIMM + h * HD + c];
  }
  // zero token padding 144..159 of Vt
  for (int i = tid; i < HD * 16; i += 256) {
    int c = i >> 4;
    Vt[c * PSTR + NT + (i & 15)] = 0;
  }
  // zero token padding 144..159 of every wave's P block
  for (int i = tid; i < 8 * 16 * 16; i += 256) {
    int wv = i >> 8, r = (i >> 4) & 15;
    Psh[wv * (16 * PSTR) + r * PSTR + NT + (i & 15)] = 0;
  }
  // window mask bits -> LDS
  for (int i = tid; i < NT * MB_W; i += 256)
    Mbsh[i] = maskb[rowbase * MB_W + i];
  __syncthreads();

  const float scale = 0.17677669529663687f;  // 1/sqrt(HD)

  for (int rb = wid; rb < 9; rb += 8) {      // 9 row-blocks of 16 query tokens
    // Q A-fragment straight from global (reused for all 9 S tiles)
    v16bf aq = load_a_frag(qkv + (rowbase + rb * 16) * QKVN + h * HD, QKVN, lane);

    // S = Q @ K^T : one 16x16x32 WMMA per column tile
    v8f acc[9];
#pragma unroll
    for (int kt = 0; kt < 9; ++kt) {
      v16bf bk = load_b_frag(&Ksh[kt * 16 * HD], HD, lane);
      v8f z = {};
      acc[kt] = __builtin_amdgcn_wmma_f32_16x16x32_bf16(false, aq, false, bk, (short)0, z, false, false);
    }

    int rlo   = rb * 16 + ((lane < 16) ? 0 : 8);
    int cbase = lane & 15;

    // scale + relative-position bias + bit-mask (in-register, C-fragment layout)
#pragma unroll
    for (int kt = 0; kt < 9; ++kt) {
      int col = kt * 16 + cbase;
      int mw  = col >> 5, mbit = col & 31;
#pragma unroll
      for (int r = 0; r < 8; ++r) {
        int row = rlo + r;
        float s = acc[kt][r] * scale + bias_full[((long long)h * NT + row) * NT + col];
        unsigned int on = (Mbsh[row * MB_W + mw] >> mbit) & 1u;
        acc[kt][r] = on ? s : -1.0e9f;
      }
    }

    // row softmax: columns of a row live in 16 lanes -> shfl_xor reduce (wave32)
    float rinv[8];
#pragma unroll
    for (int r = 0; r < 8; ++r) {
      float m = -3.4e38f;
#pragma unroll
      for (int kt = 0; kt < 9; ++kt) m = fmaxf(m, acc[kt][r]);
      for (int off = 1; off <= 8; off <<= 1) m = fmaxf(m, __shfl_xor(m, off, 32));
      float ssum = 0.f;
#pragma unroll
      for (int kt = 0; kt < 9; ++kt) {
        float e = __expf(acc[kt][r] - m);
        acc[kt][r] = e;
        ssum += e;
      }
      for (int off = 1; off <= 8; off <<= 1) ssum += __shfl_xor(ssum, off, 32);
      rinv[r] = 1.0f / ssum;
    }

    // normalized P -> LDS (bf16) in A-loadable row-major layout
    unsigned short* pbase = &Psh[wid * (16 * PSTR)];
#pragma unroll
    for (int kt = 0; kt < 9; ++kt) {
      int col = kt * 16 + cbase;
#pragma unroll
      for (int r = 0; r < 8; ++r) {
        int rl = ((lane < 16) ? 0 : 8) + r;
        pbase[rl * PSTR + col] = f2bf(acc[kt][r] * rinv[r]);
      }
    }

    // O = P @ V : 2 output tiles x 5 k-steps of 32 tokens (padded)
#pragma unroll
    for (int ntile = 0; ntile < 2; ++ntile) {
      v8f o = {};
#pragma unroll
      for (int ks = 0; ks < 5; ++ks) {
        v16bf ap = load_a_frag(pbase + ks * 32, PSTR, lane);
        v16bf bv = load_b_frag(&Vt[ntile * 16 * PSTR + ks * 32], PSTR, lane);
        o = __builtin_amdgcn_wmma_f32_16x16x32_bf16(false, ap, false, bv, (short)0, o, false, false);
      }
      int ccol = h * HD + ntile * 16 + cbase;
#pragma unroll
      for (int r = 0; r < 8; ++r) {
        long long row = rowbase + rlo + r;
        attn_out[row * DIMM + ccol] = f2bf(o[r]);
      }
    }
  }
}

// ---------------- launcher ----------------

extern "C" void kernel_launch(void* const* d_in, const int* in_sizes, int n_in,
                              void* d_out, int out_size, void* d_ws, size_t ws_size,
                              hipStream_t stream) {
  const float* x      = (const float*)d_in[0];
  const int*   mask   = (const int*)d_in[1];
  const int*   rel    = (const int*)d_in[2];
  const float* w_qkv  = (const float*)d_in[3];
  const float* b_qkv  = (const float*)d_in[4];
  const float* w_out  = (const float*)d_in[5];
  const float* b_out  = (const float*)d_in[6];
  const float* btab   = (const float*)d_in[7];
  float* out = (float*)d_out;

  char* ws = (char*)d_ws;
  size_t off = 0;
  auto alloc = [&](size_t bytes) -> void* {
    void* p = ws + off;
    off += (bytes + 255) & ~(size_t)255;
    return p;
  };

  unsigned short* x_bf    = (unsigned short*)alloc((size_t)MROWS * DIMM * 2);    //  53 MB
  unsigned short* wqkvt   = (unsigned short*)alloc((size_t)QKVN * DIMM * 2);     // 221 KB
  unsigned short* woutt   = (unsigned short*)alloc((size_t)DIMM * DIMM * 2);     //  74 KB
  unsigned short* qkv_bf  = (unsigned short*)alloc((size_t)MROWS * QKVN * 2);    // 159 MB
  unsigned short* ao_bf   = (unsigned short*)alloc((size_t)MROWS * DIMM * 2);    //  53 MB
  float*          biasf   = (float*)alloc((size_t)HEADS * NT * NT * 4);          // 498 KB
  unsigned int*   maskb   = (unsigned int*)alloc((size_t)MROWS * MB_W * 4);      // 2.7 MB
  (void)ws_size; (void)in_sizes; (void)n_in; (void)out_size;

  // 1) precompute bf16 copies, expanded bias table, mask bitfield
  {
    long long n = (long long)MROWS * DIMM;
    f32_to_bf16_kernel<<<(unsigned)((n + 255) / 256), 256, 0, stream>>>(x, x_bf, n);
  }
  transpose_to_bf16<<<(DIMM * QKVN + 255) / 256, 256, 0, stream>>>(w_qkv, wqkvt, DIMM, QKVN);
  transpose_to_bf16<<<(DIMM * DIMM + 255) / 256, 256, 0, stream>>>(w_out, woutt, DIMM, DIMM);
  bias_expand<<<(HEADS * NT * NT + 255) / 256, 256, 0, stream>>>(btab, rel, biasf);
  {
    long long n = (long long)MROWS * MB_W * 32;          // 22,118,400 (exact /256)
    mask_to_bits<<<(unsigned)(n / 256), 256, 0, stream>>>(mask, maskb);
  }

  // 2) QKV projection: (138240 x 192) @ (192 x 576), bf16 out
  {
    int tiles = (MROWS / 16) * (QKVN / 64);              // 77760, divisible by 4
    gemm_bf16<false><<<tiles / 4, 128, 0, stream>>>(x_bf, wqkvt, b_qkv, qkv_bf,
                                                    MROWS, QKVN, DIMM);
  }

  // 3) fused attention per (window, head)
  attn_kernel<<<NW * HEADS, 256, 0, stream>>>(qkv_bf, maskb, biasf, ao_bf);

  // 4) output projection: (138240 x 192) @ (192 x 192) + b_out, f32 out
  {
    int tiles = (MROWS / 16) * (DIMM / 64);              // 25920, divisible by 4
    gemm_bf16<true><<<tiles / 4, 128, 0, stream>>>(ao_bf, woutt, b_out, out,
                                                   MROWS, DIMM, DIMM);
  }
}
